// NodeHead_43490838839546
// MI455X (gfx1250) — matrix operand order repaired and checked by reading, not compiled, not run on hardware
//
#include <hip/hip_runtime.h>
#include <hip/hip_bf16.h>

typedef __attribute__((ext_vector_type(16))) __bf16 v16bf;
typedef __attribute__((ext_vector_type(8)))  float  v8f;
typedef __attribute__((ext_vector_type(4)))  unsigned int v4u;
typedef __attribute__((ext_vector_type(8)))  int  v8i_;
typedef __attribute__((ext_vector_type(4)))  int  v4i_;

#define NNODES 262144
#define BGRAPH 1024
#define NPG    256
#define LAT    256
#define HID    512
#define MT     64            // rows of the node dimension per workgroup
#define HSTR   520           // padded LDS stride (bf16) = 1040B -> bank-friendly

// LDS layout (dynamic smem; no static LDS in this kernel -> offsets from 0)
#define HBUF_BYTES   (MT * HSTR * 2)            // 66560
#define BBUF_OFF     HBUF_BYTES                 // 66560 (1024B aligned)
#define KSTEP_BYTES  32768                      // one k-step of B: 32 tiles * 1KB
#define KCHUNK       2                          // k-steps per TDM chunk
#define CHUNK_BYTES  (KCHUNK * KSTEP_BYTES)     // 65536
#define CHUNK_ELEMS  (CHUNK_BYTES / 2)          // bf16 elems
#define SMEM_BYTES   (HBUF_BYTES + 2 * CHUNK_BYTES)   // 197632 (~193KB of 320KB)

#if __has_include(<hip/amd_detail/amd_gfx1250_TDM.h>)
#define ATHENA_TDM_6ARG 1
#else
#define ATHENA_TDM_6ARG 0
#endif

static __device__ __forceinline__ float silu_f(float x) {
  return x / (1.0f + __expf(-x));
}

// ---------------------------------------------------------------------------
// TDM: DMA one contiguous row of `ndw` dwords from global -> LDS offset.
// D# per CDNA5 ISA.
// ---------------------------------------------------------------------------
static __device__ __forceinline__ void tdm_load_row(const void* gptr,
                                                    unsigned lds_off,
                                                    unsigned ndw) {
  unsigned long long ga = (unsigned long long)gptr;
  v4u g0;
  g0.x = 1u;                                            // count=1 (valid user D#)
  g0.y = lds_off;                                       // lds_addr (bytes)
  g0.z = (unsigned)(ga & 0xFFFFFFFFu);                  // global_addr[31:0]
  g0.w = (unsigned)((ga >> 32) & 0x1FFFFFFu) | (2u << 30);  // addr[56:32] | type=2
  v8i_ g1;
  g1[0] = (int)(2u << 16);                              // wg_mask=0, data_size=4B
  g1[1] = (int)((ndw & 0xFFFFu) << 16);                 // tensor_dim0[15:0]
  g1[2] = (int)(((ndw >> 16) & 0xFFFFu) | (1u << 16));  // tensor_dim0[31:16], tensor_dim1=1
  g1[3] = (int)((ndw & 0xFFFFu) << 16);                 // tile_dim0 = ndw
  g1[4] = 1;                                            // tile_dim1=1, tile_dim2=0
  g1[5] = (int)ndw;                                     // tensor_dim0_stride
  g1[6] = 0;
  g1[7] = 0;
  v4i_ z4 = {0, 0, 0, 0};
#if ATHENA_TDM_6ARG
  v8i_ z8 = {0, 0, 0, 0, 0, 0, 0, 0};
  __builtin_amdgcn_tensor_load_to_lds(g0, g1, z4, z4, z8, 0);
#else
  __builtin_amdgcn_tensor_load_to_lds(g0, g1, z4, z4, 0);
#endif
}

// ---------------------------------------------------------------------------
// Pack W0 [LAT,HID], W1 [HID,HID] (f32) into WMMA B-fragment order (bf16):
// per 32x16 KxN tile: 32 lanes x 16 contiguous-K bf16 (lane l: n=l%16,
// k=(l/16)*16+j). Tiles ordered (kt,nt) -> one k-step contiguous (TDM row).
// Also W2 transposed (f32).
// ---------------------------------------------------------------------------
__global__ void pack_w_kernel(const float* __restrict__ W0,
                              const float* __restrict__ W1,
                              const float* __restrict__ W2,
                              __bf16* __restrict__ W0p,
                              __bf16* __restrict__ W1p,
                              float* __restrict__ W2t) {
  int idx = blockIdx.x * blockDim.x + threadIdx.x;
  const int T0 = LAT * HID;          // 131072
  const int T1 = HID * HID;          // 262144
  const int T2 = HID * 3;            // 1536
  const int NT = HID / 16;           // 32 n-tiles
  if (idx < T0) {
    int tile = idx >> 9, within = idx & 511;
    int lane = within >> 4, j = within & 15;
    int kt = tile / NT, nt = tile - kt * NT;
    int n = nt * 16 + (lane & 15);
    int k = kt * 32 + (lane >> 4) * 16 + j;
    W0p[idx] = (__bf16)W0[k * HID + n];
  } else if (idx < T0 + T1) {
    int d = idx - T0;
    int tile = d >> 9, within = d & 511;
    int lane = within >> 4, j = within & 15;
    int kt = tile / NT, nt = tile - kt * NT;
    int n = nt * 16 + (lane & 15);
    int k = kt * 32 + (lane >> 4) * 16 + j;
    W1p[d] = (__bf16)W1[k * HID + n];
  } else if (idx < T0 + T1 + T2) {
    int d = idx - T0 - T1;           // d = c*512 + k
    int c = d >> 9, k = d & 511;
    W2t[d] = W2[k * 3 + c];
  }
}

// ---------------------------------------------------------------------------
// Fused 3-layer MLP per 64-row block; B double-buffered in LDS via TDM,
// 2 k-steps per chunk. 8 waves: (mblk 0..3, nhalf 0..1); each wave owns
// 16 rows x 256 cols = 16 wmma tiles. Per k-step: preload 16 B fragments
// into distinct regs, then 16 back-to-back wmmas.
// ---------------------------------------------------------------------------
__global__ __launch_bounds__(256) void mlp_fused_kernel(
    const float* __restrict__ feat,
    const __bf16* __restrict__ W0p, const float* __restrict__ b0,
    const __bf16* __restrict__ W1p, const float* __restrict__ b1,
    const float* __restrict__ W2t,  const float* __restrict__ b2,
    float* __restrict__ pred) {
  extern __shared__ char smem[];
  __bf16* hbuf = (__bf16*)smem;                 // [MT][HSTR] bf16

  const int tid   = threadIdx.x;
  const int lane  = tid & 31;
  const int wave  = tid >> 5;
  const int mblk  = wave >> 1;
  const int nhalf = wave & 1;
  const int rowA  = lane & 15;
  const int hiA   = lane >> 4;
  const int ncol  = lane & 15;

  v8f acc[16];

  // ================= GEMM0: feat[64,256] x W0[256,512] =================
  #pragma unroll
  for (int nt = 0; nt < 16; ++nt) {
    float bv = b0[nhalf * 256 + nt * 16 + ncol];
    acc[nt] = (v8f){bv, bv, bv, bv, bv, bv, bv, bv};
  }
  if (wave == 0) {
    tdm_load_row(W0p, BBUF_OFF, CHUNK_BYTES >> 2);          // chunk0 -> buf0
    __builtin_amdgcn_s_wait_tensorcnt(0);
  }
  __syncthreads();

  const float* fr = feat + ((size_t)blockIdx.x * MT + mblk * 16 + rowA) * LAT;
  const int NCH0 = (LAT / 32) / KCHUNK;                     // 4 chunks
  for (int ch = 0; ch < NCH0; ++ch) {
    const int cur = ch & 1;
    if (wave == 0 && ch + 1 < NCH0)
      tdm_load_row(W0p + (size_t)(ch + 1) * CHUNK_ELEMS,
                   BBUF_OFF + (1 - cur) * CHUNK_BYTES, CHUNK_BYTES >> 2);
    #pragma unroll
    for (int sub = 0; sub < KCHUNK; ++sub) {
      const int kt = ch * KCHUNK + sub;
      const int kb = kt * 32 + hiA * 16;
      v16bf a;
      #pragma unroll
      for (int j = 0; j < 8; ++j) {             // A layout: K kb..kb+15 contiguous
        a[j]     = (__bf16)fr[kb + j];
        a[8 + j] = (__bf16)fr[kb + 8 + j];
      }
      const v16bf* Bl = (const v16bf*)(smem + BBUF_OFF + cur * CHUNK_BYTES +
                                       sub * KSTEP_BYTES) + nhalf * 512 + lane;
      v16bf breg[16];
      #pragma unroll
      for (int i = 0; i < 16; ++i) breg[i] = Bl[i * 32];    // distinct regs: burst loads
      #pragma unroll
      for (int i = 0; i < 16; ++i)
        acc[i] = __builtin_amdgcn_wmma_f32_16x16x32_bf16(
            false, a, false, breg[i], (short)0, acc[i], false, false);
    }
    if (wave == 0) __builtin_amdgcn_s_wait_tensorcnt(0);
    __syncthreads();
  }
  #pragma unroll
  for (int nt = 0; nt < 16; ++nt) {
    #pragma unroll
    for (int r = 0; r < 8; ++r) {               // C layout: M = r + 8*hiA
      int m = mblk * 16 + r + hiA * 8;
      int c = nhalf * 256 + nt * 16 + ncol;
      hbuf[m * HSTR + c] = (__bf16)silu_f(acc[nt][r]);
    }
  }
  __syncthreads();

  // ================= GEMM1: h0[64,512] x W1[512,512] =================
  #pragma unroll
  for (int nt = 0; nt < 16; ++nt) {
    float bv = b1[nhalf * 256 + nt * 16 + ncol];
    acc[nt] = (v8f){bv, bv, bv, bv, bv, bv, bv, bv};
  }
  if (wave == 0) {
    tdm_load_row(W1p, BBUF_OFF, CHUNK_BYTES >> 2);          // chunk0 -> buf0
    __builtin_amdgcn_s_wait_tensorcnt(0);
  }
  __syncthreads();

  const __bf16* hr = hbuf + (mblk * 16 + rowA) * HSTR;      // rows 16B-aligned
  const int NCH1 = (HID / 32) / KCHUNK;                     // 8 chunks
  for (int ch = 0; ch < NCH1; ++ch) {
    const int cur = ch & 1;
    if (wave == 0 && ch + 1 < NCH1)
      tdm_load_row(W1p + (size_t)(ch + 1) * CHUNK_ELEMS,
                   BBUF_OFF + (1 - cur) * CHUNK_BYTES, CHUNK_BYTES >> 2);
    #pragma unroll
    for (int sub = 0; sub < KCHUNK; ++sub) {
      const int kt = ch * KCHUNK + sub;
      const int kb = kt * 32 + hiA * 16;
      union { uint4 u[2]; v16bf v; } ua;
      ua.u[0] = *(const uint4*)(hr + kb);                   // ds_load_b128
      ua.u[1] = *(const uint4*)(hr + kb + 8);               // ds_load_b128
      v16bf a = ua.v;
      const v16bf* Bl = (const v16bf*)(smem + BBUF_OFF + cur * CHUNK_BYTES +
                                       sub * KSTEP_BYTES) + nhalf * 512 + lane;
      v16bf breg[16];
      #pragma unroll
      for (int i = 0; i < 16; ++i) breg[i] = Bl[i * 32];
      #pragma unroll
      for (int i = 0; i < 16; ++i)
        acc[i] = __builtin_amdgcn_wmma_f32_16x16x32_bf16(
            false, a, false, breg[i], (short)0, acc[i], false, false);
    }
    if (wave == 0) __builtin_amdgcn_s_wait_tensorcnt(0);
    __syncthreads();
  }
  // all waves done reading h0 (barrier above) -> overwrite with h1
  #pragma unroll
  for (int nt = 0; nt < 16; ++nt) {
    #pragma unroll
    for (int r = 0; r < 8; ++r) {
      int m = mblk * 16 + r + hiA * 8;
      int c = nhalf * 256 + nt * 16 + ncol;
      hbuf[m * HSTR + c] = (__bf16)silu_f(acc[nt][r]);
    }
  }
  __syncthreads();

  // ================= pred = h1 @ W2 + b2 (OUT=3), vectorized =================
  if (tid < MT * 3) {
    int row = tid / 3, c = tid - row * 3;
    const uint4*  hp = (const uint4*)(hbuf + row * HSTR);   // 8 bf16 per load
    const float4* wp = (const float4*)(W2t + c * HID);      // 4 f32 per load
    float s = b2[c];
    for (int k8 = 0; k8 < HID / 8; ++k8) {
      union { uint4 u; __bf16 h[8]; } cv;
      cv.u = hp[k8];
      float4 w0 = wp[2 * k8], w1 = wp[2 * k8 + 1];
      s += (float)cv.h[0] * w0.x + (float)cv.h[1] * w0.y +
           (float)cv.h[2] * w0.z + (float)cv.h[3] * w0.w +
           (float)cv.h[4] * w1.x + (float)cv.h[5] * w1.y +
           (float)cv.h[6] * w1.z + (float)cv.h[7] * w1.w;
    }
    pred[((size_t)blockIdx.x * MT + row) * 3 + c] = s;
  }
}

// ---------------------------------------------------------------------------
// Per-graph head: mean removal, COM, torque 3x3 solve, cell==0 gating.
// ---------------------------------------------------------------------------
static __device__ __forceinline__ float wave_sum(float v) {
  #pragma unroll
  for (int off = 16; off > 0; off >>= 1) v += __shfl_down(v, off, 32);
  return v;
}

__global__ __launch_bounds__(NPG) void graph_head_kernel(
    const float* __restrict__ pred, const float* __restrict__ pos,
    const float* __restrict__ cell, const int* __restrict__ n_node,
    float* __restrict__ out) {
  const int g = blockIdx.x;
  const int tid = threadIdx.x;
  const int lane = tid & 31;
  const size_t node = (size_t)g * NPG + tid;

  float p0 = pred[node * 3 + 0], p1 = pred[node * 3 + 1], p2 = pred[node * 3 + 2];
  float x0 = pos[node * 3 + 0],  x1 = pos[node * 3 + 1],  x2 = pos[node * 3 + 2];

  __shared__ float red1[6];
  __shared__ float red2[10];
  if (tid < 6)  red1[tid] = 0.0f;
  if (tid < 10) red2[tid] = 0.0f;
  __syncthreads();

  {
    float v[6] = {p0, p1, p2, x0, x1, x2};
    #pragma unroll
    for (int i = 0; i < 6; ++i) {
      float w = wave_sum(v[i]);
      if (lane == 0) atomicAdd(&red1[i], w);     // ds_add_f32
    }
  }
  __syncthreads();

  const float inv_n = 1.0f / (float)n_node[g];
  p0 -= red1[0] * inv_n; p1 -= red1[1] * inv_n; p2 -= red1[2] * inv_n;
  const float r0 = x0 - red1[3] * inv_n;
  const float r1 = x1 - red1[4] * inv_n;
  const float r2 = x2 - red1[5] * inv_n;

  {
    float c0 = r1 * p2 - r2 * p1;
    float c1 = r2 * p0 - r0 * p2;
    float c2 = r0 * p1 - r1 * p0;
    float rr = r0 * r0 + r1 * r1 + r2 * r2;
    float v[10] = {c0, c1, c2, rr,
                   r0 * r0, r0 * r1, r0 * r2, r1 * r1, r1 * r2, r2 * r2};
    #pragma unroll
    for (int i = 0; i < 10; ++i) {
      float w = wave_sum(v[i]);
      if (lane == 0) atomicAdd(&red2[i], w);
    }
  }
  __syncthreads();

  const float t0 = red2[0], t1 = red2[1], t2 = red2[2], s = red2[3];
  const float M00 = red2[4] - s, M01 = red2[5], M02 = red2[6];
  const float M11 = red2[7] - s, M12 = red2[8], M22 = red2[9] - s;
  const float A00 = M11 * M22 - M12 * M12;
  const float A01 = M02 * M12 - M01 * M22;
  const float A02 = M01 * M12 - M02 * M11;
  const float A11 = M00 * M22 - M02 * M02;
  const float A12 = M01 * M02 - M00 * M12;
  const float A22 = M00 * M11 - M01 * M01;
  const float invdet = 1.0f / (M00 * A00 + M01 * A01 + M02 * A02);
  const float mu0 = (A00 * -t0 + A01 * -t1 + A02 * -t2) * invdet;
  const float mu1 = (A01 * -t0 + A11 * -t1 + A12 * -t2) * invdet;
  const float mu2 = (A02 * -t0 + A12 * -t1 + A22 * -t2) * invdet;

  const float* cg = cell + (size_t)g * 9;
  bool nopbc = true;
  #pragma unroll
  for (int j = 0; j < 9; ++j) nopbc = nopbc && (cg[j] == 0.0f);

  const float d0 = r1 * mu2 - r2 * mu1;
  const float d1 = r2 * mu0 - r0 * mu2;
  const float d2 = r0 * mu1 - r1 * mu0;

  out[node * 3 + 0] = nopbc ? p0 + d0 : p0;
  out[node * 3 + 1] = nopbc ? p1 + d1 : p1;
  out[node * 3 + 2] = nopbc ? p2 + d2 : p2;
}

// ---------------------------------------------------------------------------
extern "C" void kernel_launch(void* const* d_in, const int* in_sizes, int n_in,
                              void* d_out, int out_size, void* d_ws, size_t ws_size,
                              hipStream_t stream) {
  (void)in_sizes; (void)n_in; (void)out_size; (void)ws_size;
  const float* feat = (const float*)d_in[0];
  const float* pos  = (const float*)d_in[1];
  const float* cell = (const float*)d_in[2];
  const int*   nn   = (const int*)d_in[3];
  const float* W0   = (const float*)d_in[4];
  const float* b0   = (const float*)d_in[5];
  const float* W1   = (const float*)d_in[6];
  const float* b1   = (const float*)d_in[7];
  const float* W2   = (const float*)d_in[8];
  const float* b2   = (const float*)d_in[9];
  float* out = (float*)d_out;

  char* ws = (char*)d_ws;
  __bf16* W0p = (__bf16*)(ws);                         // 256 KB
  __bf16* W1p = (__bf16*)(ws + 262144);                // 512 KB
  float*  W2t = (float*)(ws + 262144 + 524288);        // 6 KB
  float*  pred = (float*)(ws + 262144 + 524288 + 8192);// 3 MB

  const int packN = LAT * HID + HID * HID + HID * 3;   // 394752 elems
  pack_w_kernel<<<(packN + 255) / 256, 256, 0, stream>>>(W0, W1, W2, W0p, W1p, W2t);

  mlp_fused_kernel<<<NNODES / MT, 256, SMEM_BYTES, stream>>>(
      feat, W0p, b0, W1p, b1, W2t, b2, pred);

  graph_head_kernel<<<BGRAPH, NPG, 0, stream>>>(pred, pos, cell, nn, out);
}
